// SelfAttention_74586402063061
// MI455X (gfx1250) — matrix operand order repaired and checked by reading, not compiled, not run on hardware
//
#include <hip/hip_runtime.h>
#include <hip/hip_bf16.h>
#include <stdint.h>

// Problem constants (from reference setup_inputs)
#define HB 4
#define HT 2048
#define HC 2048
#define NH 16
#define HD 128
#define HM (HB * HT)     // 8192 rows
#define N_QKV (3 * HC)   // 6144

typedef __attribute__((ext_vector_type(16))) __bf16 v16bf;
typedef __attribute__((ext_vector_type(8)))  float  v8f;

__device__ __forceinline__ __bf16 f2bf(float f) {
  union { float f; uint32_t u; } v; v.f = f;
  uint32_t r = v.u + 0x7fffu + ((v.u >> 16) & 1u);   // round-to-nearest-even
  union { uint16_t u; __bf16 h; } o; o.u = (uint16_t)(r >> 16);
  return o.h;
}
__device__ __forceinline__ float bf2f(__bf16 h) {
  union { uint16_t u; __bf16 h; } i; i.h = h;
  union { uint32_t u; float f; } o; o.u = ((uint32_t)i.u) << 16;
  return o.f;
}

// ---------------------------------------------------------------------------
// CDNA5 async copy: global -> LDS without a VGPR bounce (ASYNCcnt-tracked).
// GV addressing: VDST = per-lane LDS byte offset (flat addr[31:0]),
//                VADDR = per-lane 64-bit global address.
// ---------------------------------------------------------------------------
__device__ __forceinline__ void async_load_b128(const void* gsrc, void* lds_dst) {
  const uint32_t ldsoff = (uint32_t)(uintptr_t)lds_dst;  // LDS byte offset = low 32 bits of the pointer
  asm volatile("global_load_async_to_lds_b128 %0, %1, off"
               :: "v"(ldsoff), "v"((uint64_t)(uintptr_t)gsrc)
               : "memory");
}
__device__ __forceinline__ void wait_async_zero() {
  asm volatile("s_wait_asynccnt 0" ::: "memory");
}

// CDNA5 16-bit A-matrix fragment (16x32), per ISA 7.12.2:
// lanes 0-15: M=lane, VGPR0-3 hold K=0..7, VGPR4-7 hold K=16..23
// lanes 16-31: M=lane-16, VGPR0-3 hold K=8..15, VGPR4-7 hold K=24..31
__device__ __forceinline__ v16bf load_frag_a(const __bf16* base, int ld, int k0) {
  const int lane = threadIdx.x & 31;
  const int m = lane & 15, kh = lane >> 4;
  const __bf16* p = base + (size_t)m * ld + k0 + kh * 8;
  v16bf f;
#pragma unroll
  for (int i = 0; i < 8; ++i) f[i] = p[i];
#pragma unroll
  for (int i = 0; i < 8; ++i) f[8 + i] = p[16 + i];
  return f;
}

// CDNA5 16-bit B-matrix fragment (32x16), column-major source:
// element (k, n) lives at base[n*ldn + k];
// lanes 0-15: N=lane, K=0..7,16..23 ; lanes 16-31: N=lane-16, K=8..15,24..31
__device__ __forceinline__ v16bf load_frag_b(const __bf16* base, int ldn) {
  const int lane = threadIdx.x & 31;
  const int n = lane & 15, kh = lane >> 4;
  const __bf16* p = base + (size_t)n * ldn + kh * 8;
  v16bf f;
#pragma unroll
  for (int i = 0; i < 8; ++i) f[i] = p[i];
#pragma unroll
  for (int i = 0; i < 8; ++i) f[8 + i] = p[16 + i];
  return f;
}

__device__ __forceinline__ v8f wmma_bf16(v16bf a, v16bf b, v8f c) {
  return __builtin_amdgcn_wmma_f32_16x16x32_bf16(false, a, false, b, (short)0, c,
                                                 false, false);
}

// ---------------------------------------------------------------------------
// fp32 -> bf16 conversion
// ---------------------------------------------------------------------------
__global__ void __launch_bounds__(256)
cvt_f32_bf16_kernel(const float* __restrict__ in, __bf16* __restrict__ out, int n) {
  int i = blockIdx.x * blockDim.x + threadIdx.x;
  const int stride = gridDim.x * blockDim.x;
  for (; i < n; i += stride) out[i] = f2bf(in[i]);
}

// ---------------------------------------------------------------------------
// Tiled bf16 WMMA GEMM: C[M,N] = A[M,K] * B[K,N]
// Workgroup tile 128x128, BK=32; 8 waves, each wave owns a 32x64 subtile.
// A tile staged via async global->LDS copies; B tile transposed via registers.
// ---------------------------------------------------------------------------
template <bool OUT_BF16>
__global__ void __launch_bounds__(256)
gemm_bf16_wmma(const __bf16* __restrict__ A, const __bf16* __restrict__ B,
               void* __restrict__ Cout, int M, int N, int K) {
  __shared__ __attribute__((aligned(16))) __bf16 As[128 * 32];  // row-major [m][k]
  __shared__ __attribute__((aligned(16))) __bf16 Bs[128 * 32];  // col-major [n][k]

  const int tid = threadIdx.x;
  const int lane = tid & 31;
  const int w = tid >> 5;       // 0..7
  const int wm = w >> 1;        // 0..3 : M offset wm*32
  const int wn = w & 1;         // 0..1 : N offset wn*64
  const int m0 = blockIdx.y * 128;
  const int n0 = blockIdx.x * 128;

  const v8f zero = {};
  v8f acc[2][4];
#pragma unroll
  for (int i = 0; i < 2; ++i)
#pragma unroll
    for (int j = 0; j < 4; ++j) acc[i][j] = zero;

  const int arow = tid >> 1, ahalf = tid & 1;   // A staging: 128 rows x 32 k
  const int bk = tid >> 3,  bcg = tid & 7;      // B staging: 32 k x 128 n

  const int KT = K >> 5;
  for (int kt = 0; kt < KT; ++kt) {
    const int k0 = kt << 5;
    // stage A tile: async global->LDS (no VGPR bounce, ASYNCcnt-tracked)
    {
      const __bf16* src = A + (size_t)(m0 + arow) * K + k0 + ahalf * 16;
      __bf16* dst = As + arow * 32 + ahalf * 16;
      async_load_b128(src, dst);
      async_load_b128(src + 8, dst + 8);
    }
    // stage B tile, transposing to column-major for contiguous B-fragments
    {
      const __bf16* src = B + (size_t)(k0 + bk) * N + n0 + bcg * 16;
      union { uint4 v; __bf16 h[8]; } u0, u1;
      u0.v = ((const uint4*)src)[0];
      u1.v = ((const uint4*)src)[1];
#pragma unroll
      for (int i = 0; i < 8; ++i) {
        Bs[(bcg * 16 + i) * 32 + bk]     = u0.h[i];
        Bs[(bcg * 16 + 8 + i) * 32 + bk] = u1.h[i];
      }
    }
    if (kt + 1 < KT) {  // pull next K-tile toward L2 (global_prefetch_b8)
      __builtin_prefetch(A + (size_t)(m0 + arow) * K + k0 + 32, 0, 1);
      __builtin_prefetch(B + (size_t)(k0 + 32 + bk) * N + n0 + bcg * 16, 0, 1);
    }
    wait_async_zero();   // this wave's async copies landed in LDS
    __syncthreads();

    v16bf af[2], bfr[4];
#pragma unroll
    for (int i = 0; i < 2; ++i)
      af[i] = load_frag_a(As + (wm * 32 + i * 16) * 32, 32, 0);
#pragma unroll
    for (int j = 0; j < 4; ++j)
      bfr[j] = load_frag_b(Bs + (wn * 64 + j * 16) * 32, 32);
#pragma unroll
    for (int i = 0; i < 2; ++i)
#pragma unroll
      for (int j = 0; j < 4; ++j) acc[i][j] = wmma_bf16(af[i], bfr[j], acc[i][j]);
    __syncthreads();
  }

  // epilogue: C/D layout row = r + 8*(lane>=16), col = lane&15
  const int rh = lane >> 4, cn = lane & 15;
#pragma unroll
  for (int i = 0; i < 2; ++i)
#pragma unroll
    for (int j = 0; j < 4; ++j)
#pragma unroll
      for (int r = 0; r < 8; ++r) {
        const int row = m0 + wm * 32 + i * 16 + r + 8 * rh;
        const int col = n0 + wn * 64 + j * 16 + cn;
        const float v = acc[i][j][r];
        if (OUT_BF16) ((__bf16*)Cout)[(size_t)row * N + col] = f2bf(v);
        else          ((float*)Cout)[(size_t)row * N + col] = v;
      }
}

// ---------------------------------------------------------------------------
// RoPE + repack: qkv (B*T, 3C) bf16 -> Q[bh][t][d], K[bh][t][d], V^T[bh][d][t]
// ---------------------------------------------------------------------------
__global__ void __launch_bounds__(128)
rope_pack_kernel(const __bf16* __restrict__ qkv, __bf16* __restrict__ Qd,
                 __bf16* __restrict__ Kd, __bf16* __restrict__ VTd) {
  const int t = blockIdx.x;
  const int bh = blockIdx.y;           // b*NH + h
  const int h = bh & (NH - 1);
  const int b = bh >> 4;
  const int d = threadIdx.x;           // 0..127
  const size_t row = (size_t)(b * HT + t) * N_QKV;
  const size_t qk_off = (size_t)h * HD;

  // V transposed: [bh][d][t] so PV B-fragments are contiguous in t
  VTd[((size_t)bh * HD + d) * HT + t] = qkv[row + 2 * HC + qk_off + d];

  if (d < HD / 2) {
    const float inv = __powf(10000.0f, -(float)d / (float)(HD / 2));
    float s, c;
    __sincosf((float)t * inv, &s, &c);
    const size_t qdst = ((size_t)bh * HT + t) * HD;
    {
      const float x1 = bf2f(qkv[row + qk_off + d]);
      const float x2 = bf2f(qkv[row + qk_off + d + 64]);
      Qd[qdst + d]      = f2bf(x1 * c - x2 * s);
      Qd[qdst + d + 64] = f2bf(x2 * c + x1 * s);
    }
    {
      const float x1 = bf2f(qkv[row + HC + qk_off + d]);
      const float x2 = bf2f(qkv[row + HC + qk_off + d + 64]);
      Kd[qdst + d]      = f2bf(x1 * c - x2 * s);
      Kd[qdst + d + 64] = f2bf(x2 * c + x1 * s);
    }
  }
}

// ---------------------------------------------------------------------------
// Causal flash attention, one wave per (b,h, 16-row query tile).
// QK^T: 8 WMMAs / 32-key block; PV: 8 WMMAs / block; P re-laid-out via LDS.
// ---------------------------------------------------------------------------
__global__ void __launch_bounds__(32)
flash_attn_kernel(const __bf16* __restrict__ Q, const __bf16* __restrict__ Kd,
                  const __bf16* __restrict__ VT, __bf16* __restrict__ O) {
  __shared__ __attribute__((aligned(16))) __bf16 Pt[16 * 32];
  const int lane = threadIdx.x & 31;
  const int q0 = blockIdx.x * 16;
  const int bh = blockIdx.y;
  const int h = bh & (NH - 1);
  const int b = bh >> 4;
  const float scale = 0.08838834764831845f;  // 1/sqrt(128)
  const float NEG = -3.0e38f;

  const __bf16* Qh = Q  + (size_t)bh * HT * HD;
  const __bf16* Kh = Kd + (size_t)bh * HT * HD;
  const __bf16* Vh = VT + (size_t)bh * HD * HT;

  v16bf qf[4];
#pragma unroll
  for (int kk = 0; kk < 4; ++kk)
    qf[kk] = load_frag_a(Qh + (size_t)q0 * HD, HD, kk * 32);

  const v8f zero = {};
  v8f o[8];
#pragma unroll
  for (int j = 0; j < 8; ++j) o[j] = zero;
  float m[8], l[8];
#pragma unroll
  for (int r = 0; r < 8; ++r) { m[r] = NEG; l[r] = 0.0f; }

  const int rh = lane >> 4, cn = lane & 15;
  const int nkb = (q0 + 16 + 31) >> 5;
  for (int kb = 0; kb < nkb; ++kb) {
    const int key0 = kb << 5;

    // S = scale * Q K^T  (16 x 32), with causal mask
    v8f s[2];
#pragma unroll
    for (int half = 0; half < 2; ++half) {
      v8f acc = zero;
#pragma unroll
      for (int kk = 0; kk < 4; ++kk) {
        v16bf kf = load_frag_b(Kh + (size_t)(key0 + half * 16) * HD + kk * 32, HD);
        acc = wmma_bf16(qf[kk], kf, acc);
      }
      const int key = key0 + half * 16 + cn;
#pragma unroll
      for (int r = 0; r < 8; ++r) {
        const int qrow = q0 + r + 8 * rh;
        acc[r] = (key <= qrow) ? acc[r] * scale : NEG;
      }
      s[half] = acc;
    }

    // online softmax: rows reduce across 16 lanes of each half-wave
    float mn[8], alpha[8];
#pragma unroll
    for (int r = 0; r < 8; ++r) {
      float rm = fmaxf(s[0][r], s[1][r]);
#pragma unroll
      for (int off = 8; off > 0; off >>= 1)
        rm = fmaxf(rm, __shfl_xor(rm, off, 16));
      mn[r] = fmaxf(m[r], rm);
      alpha[r] = __expf(m[r] - mn[r]);
      m[r] = mn[r];
    }
#pragma unroll
    for (int half = 0; half < 2; ++half)
#pragma unroll
      for (int r = 0; r < 8; ++r) s[half][r] = __expf(s[half][r] - mn[r]);
#pragma unroll
    for (int r = 0; r < 8; ++r) {
      float rs = s[0][r] + s[1][r];
#pragma unroll
      for (int off = 8; off > 0; off >>= 1)
        rs += __shfl_xor(rs, off, 16);
      l[r] = l[r] * alpha[r] + rs;
#pragma unroll
      for (int j = 0; j < 8; ++j) o[j][r] *= alpha[r];
    }

    // P (C/D layout) -> LDS row-major 16x32 -> reload as A-fragment
#pragma unroll
    for (int r = 0; r < 8; ++r) {
      Pt[(r + 8 * rh) * 32 + cn]      = f2bf(s[0][r]);
      Pt[(r + 8 * rh) * 32 + 16 + cn] = f2bf(s[1][r]);
    }
    asm volatile("s_wait_dscnt 0" ::: "memory");
    v16bf pf = load_frag_a(Pt, 32, 0);

    // O += P V  (V^T layout gives contiguous B-fragments)
#pragma unroll
    for (int j = 0; j < 8; ++j) {
      v16bf vf = load_frag_b(Vh + (size_t)(j * 16) * HT + key0, HT);
      o[j] = wmma_bf16(pf, vf, o[j]);
    }
  }

  // normalize and scatter into (B*T, C) bf16 buffer for the output projection
#pragma unroll
  for (int j = 0; j < 8; ++j)
#pragma unroll
    for (int r = 0; r < 8; ++r) {
      const int row = q0 + r + 8 * rh;
      const int col = h * HD + j * 16 + cn;
      O[((size_t)b * HT + row) * HC + col] = f2bf(o[j][r] / l[r]);
    }
}

// ---------------------------------------------------------------------------
extern "C" void kernel_launch(void* const* d_in, const int* in_sizes, int n_in,
                              void* d_out, int out_size, void* d_ws, size_t ws_size,
                              hipStream_t stream) {
  const float* x     = (const float*)d_in[0];
  const float* Wqkv  = (const float*)d_in[1];
  const float* Wproj = (const float*)d_in[2];

  char* ws = (char*)d_ws;
  size_t off = 0;
  auto alloc = [&](size_t bytes) {
    char* p = ws + off;
    off += (bytes + 255) & ~(size_t)255;
    return p;
  };
  __bf16* x_bf     = (__bf16*)alloc((size_t)HM * HC * 2);
  __bf16* wqkv_bf  = (__bf16*)alloc((size_t)HC * N_QKV * 2);
  __bf16* wproj_bf = (__bf16*)alloc((size_t)HC * HC * 2);
  __bf16* qkv_bf   = (__bf16*)alloc((size_t)HM * N_QKV * 2);
  __bf16* q_bf     = (__bf16*)alloc((size_t)HB * NH * HT * HD * 2);
  __bf16* k_bf     = (__bf16*)alloc((size_t)HB * NH * HT * HD * 2);
  __bf16* vT_bf    = (__bf16*)alloc((size_t)HB * NH * HD * HT * 2);
  __bf16* attn_bf  = (__bf16*)alloc((size_t)HM * HC * 2);

  cvt_f32_bf16_kernel<<<4096, 256, 0, stream>>>(x, x_bf, HM * HC);
  cvt_f32_bf16_kernel<<<4096, 256, 0, stream>>>(Wqkv, wqkv_bf, HC * N_QKV);
  cvt_f32_bf16_kernel<<<4096, 256, 0, stream>>>(Wproj, wproj_bf, HC * HC);

  // QKV projection: (8192 x 2048) @ (2048 x 6144) -> bf16
  gemm_bf16_wmma<true><<<dim3(N_QKV / 128, HM / 128), 256, 0, stream>>>(
      x_bf, wqkv_bf, qkv_bf, HM, N_QKV, HC);

  // RoPE + head repack (+ V transpose)
  rope_pack_kernel<<<dim3(HT, HB * NH), 128, 0, stream>>>(qkv_bf, q_bf, k_bf, vT_bf);

  // causal flash attention
  flash_attn_kernel<<<dim3(HT / 16, HB * NH), 32, 0, stream>>>(q_bf, k_bf, vT_bf,
                                                               attn_bf);

  // output projection: (8192 x 2048) @ (2048 x 2048) -> fp32 d_out
  gemm_bf16_wmma<false><<<dim3(HC / 128, HM / 128), 256, 0, stream>>>(
      attn_bf, wproj_bf, d_out, HM, HC, HC);
}
